// Mamba_35613868818666
// MI455X (gfx1250) — matrix operand order repaired and checked by reading, not compiled, not run on hardware
//
#include <hip/hip_runtime.h>
#include <hip/hip_bf16.h>
#include <cstdint>

typedef _Float16 half_t;
typedef __attribute__((ext_vector_type(16))) _Float16 v16h;
typedef __attribute__((ext_vector_type(8)))  float    v8f;

// b128 async payload type: int __vector(4), with explicit address spaces
typedef int v4i __attribute__((__vector_size__(4 * sizeof(int))));
typedef __attribute__((address_space(1))) v4i glb_v4i;
typedef __attribute__((address_space(3))) v4i lds_v4i;
typedef __attribute__((address_space(3))) void lds_void;

#define NBATCH 4
#define L0     4096
#define BL0    (NBATCH*L0)
#define SCH    128      // scan chunk length

__device__ __forceinline__ float siluf(float x) { return x / (1.f + __expf(-x)); }

// CDNA5 async global->LDS copy (16B per lane), tracked by ASYNCcnt.
__device__ __forceinline__ void async_load_b128(const void* g, void* l)
{
#if __has_builtin(__builtin_amdgcn_global_load_async_to_lds_b128)
    __builtin_amdgcn_global_load_async_to_lds_b128(
        (glb_v4i*)g, (lds_v4i*)l, 0, 0);
#else
    unsigned lofs = (unsigned)(uintptr_t)(lds_void*)l;
    asm volatile("global_load_async_to_lds_b128 %0, %1, off"
                 :: "v"(lofs), "v"((unsigned long long)(uintptr_t)g)
                 : "memory");
#endif
}

__device__ __forceinline__ void wait_async0()
{
#if __has_builtin(__builtin_amdgcn_s_wait_asynccnt)
    __builtin_amdgcn_s_wait_asynccnt(0);
#else
    asm volatile("s_wait_asynccnt 0" ::: "memory");
#endif
}

// ---------------------------------------------------------------------------
// Generic tiled WMMA GEMM:  out[M,N] = A[M,K] * Bw[N,K]^T  (+ epilogues)
//   block tile 64(M) x 128(N), 4 waves (2x2), per-wave 32x64 => 8 wmma / K-step
//   Double-buffered LDS tiles filled with GLOBAL_LOAD_ASYNC_TO_LDS_B128,
//   paced by s_wait_asynccnt + one workgroup barrier per K-step.
//   EPI: 0 = plain, 1 = softplus(acc + bias[n]), 2 = resid + scale[n]*acc
//   ACCUM: pre-add existing out (used for multi-tap conv-as-GEMM)
//   rowShift: A row shift with zero fill at sequence boundaries (emb conv taps)
// ---------------------------------------------------------------------------
template<typename TA, int EPI, bool ACCUM>
__global__ __launch_bounds__(128) void k_gemm(
    const TA* __restrict__ A, const half_t* __restrict__ Bw,
    float* __restrict__ out, int M, int N, int K,
    int lda, int ldo,
    const float* __restrict__ bias, const float* __restrict__ scale,
    const float* __restrict__ resid,
    int rowShift, int Lseq)
{
    __shared__ __align__(16) half_t sA[2][64 * 32];
    __shared__ __align__(16) half_t sB[2][128 * 32];

    const int tid  = threadIdx.x;
    const int lane = tid & 31;
    const int wid  = tid >> 5;
    const int waveM = wid >> 1;      // 0..1
    const int waveN = wid & 1;       // 0..1
    const int m0 = blockIdx.y * 64;
    const int n0 = blockIdx.x * 128;
    const int rhalf = lane >> 4;
    const int r     = lane & 15;

    auto stageA = [&](int k0, int buf) {
        #pragma unroll
        for (int it = 0; it < 2; ++it) {
            int idx = tid + it * 128;           // 256 chunks of 8 halfs
            int row = idx >> 2;
            int cc  = (idx & 3) * 8;
            int m   = m0 + row;
            bool valid = (m < M);
            if (valid && rowShift != 0) {
                int l  = m % Lseq;
                int ls = l + rowShift;
                valid = (ls >= 0) && (ls < Lseq);
            }
            half_t* dst = &sA[buf][row * 32 + cc];
            if constexpr (sizeof(TA) == 2) {
                if (valid) {
                    async_load_b128(A + (size_t)(m + rowShift) * lda + (k0 + cc), dst);
                } else {
                    *(uint4*)dst = make_uint4(0u, 0u, 0u, 0u);
                }
            } else {
                if (valid) {
                    const TA* src = A + (size_t)(m + rowShift) * lda + (k0 + cc);
                    #pragma unroll
                    for (int j = 0; j < 8; ++j) dst[j] = (half_t)src[j];
                } else {
                    *(uint4*)dst = make_uint4(0u, 0u, 0u, 0u);
                }
            }
        }
    };
    auto stageB = [&](int k0, int buf) {
        #pragma unroll
        for (int it = 0; it < 4; ++it) {
            int idx = tid + it * 128;           // 512 chunks of 8 halfs
            int row = idx >> 2;
            int cc  = (idx & 3) * 8;
            int n   = n0 + row;
            half_t* dst = &sB[buf][row * 32 + cc];
            if (n < N) {
                async_load_b128(Bw + (size_t)n * K + (k0 + cc), dst);
            } else {
                *(uint4*)dst = make_uint4(0u, 0u, 0u, 0u);
            }
        }
    };

    v8f acc[2][4] = {};

    stageA(0, 0);
    stageB(0, 0);
    const int nk = K >> 5;
    for (int ki = 0; ki < nk; ++ki) {
        const int buf = ki & 1;
        wait_async0();          // this wave's fills of sA/sB[buf] landed in LDS
        __syncthreads();        // every wave's fills of [buf] visible
        if (ki + 1 < nk) {      // prefetch next K-slab into the other buffer
            stageA((ki + 1) * 32, buf ^ 1);
            stageB((ki + 1) * 32, buf ^ 1);
        }

        // ---- build fragments (CDNA5 16x16x32 f16 layouts) ----
        v16h afrag[2];
        #pragma unroll
        for (int ms = 0; ms < 2; ++ms) {
            int rowL = waveM * 32 + ms * 16 + r;
            #pragma unroll
            for (int p = 0; p < 8; ++p) {
                int kk = (p >> 2) * 16 + (p & 3) * 2 + rhalf * 8;
                afrag[ms][2 * p]     = sA[buf][rowL * 32 + kk];
                afrag[ms][2 * p + 1] = sA[buf][rowL * 32 + kk + 1];
            }
        }
        v16h bfrag[4];
        #pragma unroll
        for (int ns = 0; ns < 4; ++ns) {
            int colL = waveN * 64 + ns * 16 + r;
            #pragma unroll
            for (int v = 0; v < 8; ++v) {
                int kk = (v >> 2) * 16 + (v & 3) * 2 + rhalf * 8;
                bfrag[ns][2 * v]     = sB[buf][colL * 32 + kk];
                bfrag[ns][2 * v + 1] = sB[buf][colL * 32 + kk + 1];
            }
        }
        #pragma unroll
        for (int ms = 0; ms < 2; ++ms)
            #pragma unroll
            for (int ns = 0; ns < 4; ++ns)
                acc[ms][ns] = __builtin_amdgcn_wmma_f32_16x16x32_f16(
                    false, afrag[ms], false, bfrag[ns],
                    (short)0, acc[ms][ns], false, false);
    }

    // ---- epilogue ----
    #pragma unroll
    for (int ms = 0; ms < 2; ++ms)
        #pragma unroll
        for (int ns = 0; ns < 4; ++ns)
            #pragma unroll
            for (int v = 0; v < 8; ++v) {
                int m = m0 + waveM * 32 + ms * 16 + rhalf * 8 + v;
                int n = n0 + waveN * 64 + ns * 16 + r;
                if (m < M && n < N) {
                    float a = acc[ms][ns][v];
                    size_t o = (size_t)m * ldo + n;
                    if constexpr (ACCUM) a += out[o];
                    if constexpr (EPI == 1) {
                        float xv = a + bias[n];
                        a = (xv > 20.f) ? xv : __logf(1.f + __expf(xv));
                    } else if constexpr (EPI == 2) {
                        a = resid[o] + scale[n] * a;
                    }
                    out[o] = a;
                }
            }
}

// ---------------------------------------------------------------------------
// LayerNorm over 512 channels (token-major), one wave per token.
// ---------------------------------------------------------------------------
template<bool RELU, bool WF32, bool WF16>
__global__ __launch_bounds__(256) void k_ln(
    const float* __restrict__ in, const float* __restrict__ w,
    const float* __restrict__ bb, float* __restrict__ of32,
    half_t* __restrict__ of16, int ntok)
{
    int wid = threadIdx.x >> 5, lane = threadIdx.x & 31;
    int tok = blockIdx.x * 8 + wid;
    if (tok >= ntok) return;
    const float* row = in + (size_t)tok * 512;
    float v[16], sum = 0.f;
    #pragma unroll
    for (int j = 0; j < 16; ++j) { v[j] = row[lane + j * 32]; sum += v[j]; }
    #pragma unroll
    for (int off = 16; off > 0; off >>= 1) sum += __shfl_xor(sum, off, 32);
    float mu = sum * (1.f / 512.f);
    float vs = 0.f;
    #pragma unroll
    for (int j = 0; j < 16; ++j) { float d = v[j] - mu; vs += d * d; }
    #pragma unroll
    for (int off = 16; off > 0; off >>= 1) vs += __shfl_xor(vs, off, 32);
    float rstd = rsqrtf(vs * (1.f / 512.f) + 1e-5f);
    #pragma unroll
    for (int j = 0; j < 16; ++j) {
        int c = lane + j * 32;
        float o = (v[j] - mu) * rstd * w[c] + bb[c];
        if (RELU) o = fmaxf(o, 0.f);
        if constexpr (WF32) of32[(size_t)tok * 512 + c] = o;
        if constexpr (WF16) of16[(size_t)tok * 512 + c] = (half_t)o;
    }
}

// FPN head: channel LayerNorm + transpose to channel-first output layout
__global__ __launch_bounds__(256) void k_fpn(
    const float* __restrict__ in, const float* __restrict__ w,
    const float* __restrict__ bb, float* __restrict__ outp,
    int Lf, int ntok)
{
    int wid = threadIdx.x >> 5, lane = threadIdx.x & 31;
    int tok = blockIdx.x * 8 + wid;
    if (tok >= ntok) return;
    int b = tok / Lf, l = tok % Lf;
    const float* row = in + (size_t)tok * 512;
    float v[16], sum = 0.f;
    #pragma unroll
    for (int j = 0; j < 16; ++j) { v[j] = row[lane + j * 32]; sum += v[j]; }
    #pragma unroll
    for (int off = 16; off > 0; off >>= 1) sum += __shfl_xor(sum, off, 32);
    float mu = sum * (1.f / 512.f);
    float vs = 0.f;
    #pragma unroll
    for (int j = 0; j < 16; ++j) { float d = v[j] - mu; vs += d * d; }
    #pragma unroll
    for (int off = 16; off > 0; off >>= 1) vs += __shfl_xor(vs, off, 32);
    float rstd = rsqrtf(vs * (1.f / 512.f) + 1e-5f);
    #pragma unroll
    for (int j = 0; j < 16; ++j) {
        int c = lane + j * 32;
        float o = (v[j] - mu) * rstd * w[c] + bb[c];
        outp[((size_t)(b * 512 + c)) * Lf + l] = o;
    }
}

// Depthwise causal conv (K=4) + SiLU; rev maps logical time -> physical pos.
__global__ __launch_bounds__(256) void k_dwconv(
    const float* __restrict__ xz, const float* __restrict__ w,
    const float* __restrict__ bias, float* __restrict__ xc,
    half_t* __restrict__ xch, int Lcur, int rev)
{
    int b  = blockIdx.z;
    int t0 = blockIdx.y * 64;
    int e  = blockIdx.x * 256 + threadIdx.x;
    float w0 = w[e * 4], w1 = w[e * 4 + 1], w2 = w[e * 4 + 2], w3 = w[e * 4 + 3];
    float bv = bias[e];
    auto ld = [&](int tlog) -> float {
        if (tlog < 0) return 0.f;
        int lp = rev ? (Lcur - 1 - tlog) : tlog;
        return xz[((size_t)(b * Lcur + lp)) * 2048 + e];
    };
    float x3 = ld(t0 - 3), x2 = ld(t0 - 2), x1 = ld(t0 - 1);
    for (int tt = 0; tt < 64; ++tt) {
        int tlog = t0 + tt;
        int lp = rev ? (Lcur - 1 - tlog) : tlog;
        float xt = xz[((size_t)(b * Lcur + lp)) * 2048 + e];
        float a  = w0 * x3 + w1 * x2 + w2 * x1 + w3 * xt + bv;
        float s  = siluf(a);
        size_t o = ((size_t)(b * Lcur + lp)) * 1024 + e;
        xc[o]  = s;
        xch[o] = (half_t)s;
        x3 = x2; x2 = x1; x1 = xt;
    }
}

// Selective scan pass 1: per-chunk cumulative dA product P and local state S.
__global__ __launch_bounds__(256) void k_scan1(
    const float* __restrict__ dt, const float* __restrict__ xc,
    const float* __restrict__ proj, const float* __restrict__ A_log,
    float* __restrict__ P, float* __restrict__ S,
    int Lcur, int NC, int rev)
{
    __shared__ float sB[SCH][16];
    int b = blockIdx.z, ch = blockIdx.y;
    int d = blockIdx.x * 256 + threadIdx.x;
    int t0 = ch * SCH;
    for (int i = threadIdx.x; i < SCH * 16; i += 256) {
        int tl = i >> 4, n = i & 15;
        int tlog = t0 + tl;
        int lp = rev ? (Lcur - 1 - tlog) : tlog;
        sB[tl][n] = proj[((size_t)(b * Lcur + lp)) * 64 + 32 + n];
    }
    __syncthreads();
    float Av[16], p[16], s[16];
    #pragma unroll
    for (int n = 0; n < 16; ++n) {
        Av[n] = -__expf(A_log[d * 16 + n]);
        p[n] = 1.f; s[n] = 0.f;
    }
    for (int tl = 0; tl < SCH; ++tl) {
        int tlog = t0 + tl;
        int lp = rev ? (Lcur - 1 - tlog) : tlog;
        size_t tok = ((size_t)(b * Lcur + lp)) * 1024 + d;
        float dtv = dt[tok];
        float du  = dtv * xc[tok];
        #pragma unroll
        for (int n = 0; n < 16; ++n) {
            float dA = __expf(dtv * Av[n]);
            p[n] *= dA;
            s[n] = dA * s[n] + du * sB[tl][n];
        }
    }
    size_t basei = (((size_t)b * 1024 + d) * NC + ch) * 16;
    #pragma unroll
    for (int n = 0; n < 16; ++n) { P[basei + n] = p[n]; S[basei + n] = s[n]; }
}

// Pass 2: sequential prefix over chunks (parallel over b*d*n).
__global__ __launch_bounds__(256) void k_scan2(
    const float* __restrict__ P, const float* __restrict__ S,
    float* __restrict__ H0, int NC)
{
    int i = blockIdx.x * 256 + threadIdx.x;   // b*1024*16 threads
    int n = i & 15;
    size_t bd = (size_t)(i >> 4);
    float h = 0.f;
    size_t basei = bd * NC * 16 + n;
    for (int c = 0; c < NC; ++c) {
        H0[basei + (size_t)c * 16] = h;
        h = P[basei + (size_t)c * 16] * h + S[basei + (size_t)c * 16];
    }
}

// Pass 3: rescan with correct initial state, emit y (+ D*u), accumulate dirs.
__global__ __launch_bounds__(256) void k_scan3(
    const float* __restrict__ dt, const float* __restrict__ xc,
    const float* __restrict__ proj, const float* __restrict__ A_log,
    const float* __restrict__ H0, const float* __restrict__ Dp,
    float* __restrict__ ybuf, int Lcur, int NC, int rev, int addout)
{
    __shared__ float sB[SCH][16];
    __shared__ float sC[SCH][16];
    int b = blockIdx.z, ch = blockIdx.y;
    int d = blockIdx.x * 256 + threadIdx.x;
    int t0 = ch * SCH;
    for (int i = threadIdx.x; i < SCH * 16; i += 256) {
        int tl = i >> 4, n = i & 15;
        int tlog = t0 + tl;
        int lp = rev ? (Lcur - 1 - tlog) : tlog;
        size_t pbase = ((size_t)(b * Lcur + lp)) * 64;
        sB[tl][n] = proj[pbase + 32 + n];
        sC[tl][n] = proj[pbase + 48 + n];
    }
    __syncthreads();
    float Av[16], h[16];
    size_t hbase = (((size_t)b * 1024 + d) * NC + ch) * 16;
    #pragma unroll
    for (int n = 0; n < 16; ++n) {
        Av[n] = -__expf(A_log[d * 16 + n]);
        h[n] = H0[hbase + n];
    }
    float Dv = Dp[d];
    for (int tl = 0; tl < SCH; ++tl) {
        int tlog = t0 + tl;
        int lp = rev ? (Lcur - 1 - tlog) : tlog;
        size_t tok = ((size_t)(b * Lcur + lp)) * 1024 + d;
        float dtv = dt[tok];
        float u   = xc[tok];
        float du  = dtv * u;
        float y = 0.f;
        #pragma unroll
        for (int n = 0; n < 16; ++n) {
            float dA = __expf(dtv * Av[n]);
            h[n] = dA * h[n] + du * sB[tl][n];
            y += h[n] * sC[tl][n];
        }
        float outv = y + Dv * u;
        if (addout) ybuf[tok] += outv; else ybuf[tok] = outv;
    }
}

// Gating: yg = (y_f + y_b) * silu(z), converted to f16 for out_proj GEMM.
__global__ void k_gate(const float* __restrict__ xz, const float* __restrict__ ybuf,
                       half_t* __restrict__ ygh, size_t total)
{
    size_t i = (size_t)blockIdx.x * 256 + threadIdx.x;
    if (i >= total) return;
    size_t tok = i >> 10;
    int d = (int)(i & 1023);
    float z = xz[tok * 2048 + 1024 + d];
    ygh[i] = (half_t)(ybuf[i] * siluf(z));
}

// MaxPool window 3, stride 2, pad (1,1), token-major.
__global__ void k_pool(const float* __restrict__ in, float* __restrict__ outp,
                       int Lin, int Lout, size_t total)
{
    size_t i = (size_t)blockIdx.x * 256 + threadIdx.x;
    if (i >= total) return;
    int d  = (int)(i % 512);
    int lo = (int)((i / 512) % Lout);
    int b  = (int)(i / ((size_t)512 * Lout));
    float m = -3.4028235e38f;
    #pragma unroll
    for (int k = -1; k <= 1; ++k) {
        int li = 2 * lo + k;
        if (li >= 0 && li < Lin)
            m = fmaxf(m, in[((size_t)(b * Lin + li)) * 512 + d]);
    }
    outp[((size_t)(b * Lout + lo)) * 512 + d] = m;
}

__global__ void k_f32tof16(const float* __restrict__ in, half_t* __restrict__ outp, size_t n)
{
    size_t i = (size_t)blockIdx.x * 256 + threadIdx.x;
    if (i < n) outp[i] = (half_t)in[i];
}

// x (B,512,L) channel-first -> token-major f16
__global__ void k_xpose(const float* __restrict__ x, half_t* __restrict__ xh, size_t n)
{
    size_t i = (size_t)blockIdx.x * 256 + threadIdx.x;
    if (i >= n) return;
    int l = (int)(i % L0);
    size_t r = i / L0;
    int c = (int)(r % 512);
    int b = (int)(r / 512);
    xh[((size_t)(b * L0 + l)) * 512 + c] = (half_t)x[i];
}

// emb_conv_w (2,512,512,3) -> per-tap [layer][k][o][c] f16
__global__ void k_embw(const float* __restrict__ w, half_t* __restrict__ outp, size_t n)
{
    size_t i = (size_t)blockIdx.x * 256 + threadIdx.x;
    if (i >= n) return;
    int k = (int)(i % 3);
    size_t r = i / 3;
    int c = (int)(r % 512); r /= 512;
    int o = (int)(r % 512);
    int layer = (int)(r / 512);
    outp[(((size_t)layer * 3 + k) * 512 + o) * 512 + c] = (half_t)w[i];
}

// ---------------------------------------------------------------------------
extern "C" void kernel_launch(void* const* d_in, const int* in_sizes, int n_in,
                              void* d_out, int out_size, void* d_ws, size_t ws_size,
                              hipStream_t stream)
{
    const float* x          = (const float*)d_in[0];
    const float* emb_conv_w = (const float*)d_in[1];
    const float* emb_ln_w   = (const float*)d_in[2];
    const float* emb_ln_b   = (const float*)d_in[3];
    const float* blk_ln_w   = (const float*)d_in[4];
    const float* blk_ln_b   = (const float*)d_in[5];
    const float* in_proj_w  = (const float*)d_in[6];
    const float* conv_f_w   = (const float*)d_in[7];
    const float* conv_f_b   = (const float*)d_in[8];
    const float* conv_b_w   = (const float*)d_in[9];
    const float* conv_b_b   = (const float*)d_in[10];
    const float* x_proj_f_w = (const float*)d_in[11];
    const float* x_proj_b_w = (const float*)d_in[12];
    const float* dt_proj_f_w= (const float*)d_in[13];
    const float* dt_proj_f_b= (const float*)d_in[14];
    const float* dt_proj_b_w= (const float*)d_in[15];
    const float* dt_proj_b_b= (const float*)d_in[16];
    const float* A_log_f    = (const float*)d_in[17];
    const float* A_log_b    = (const float*)d_in[18];
    const float* D_f        = (const float*)d_in[19];
    const float* D_b        = (const float*)d_in[20];
    const float* out_proj_w = (const float*)d_in[21];
    const float* affine     = (const float*)d_in[22];
    const float* fpn_ln_w   = (const float*)d_in[23];
    const float* fpn_ln_b   = (const float*)d_in[24];
    float* outp = (float*)d_out;
    (void)in_sizes; (void)n_in; (void)out_size; (void)ws_size;

    char* base = (char*)d_ws;
    size_t off = 0;
    auto alloc = [&](size_t bytes) -> char* {
        char* p = base + off;
        off += (bytes + 255) & ~(size_t)255;
        return p;
    };
    half_t* aTok = (half_t*)alloc((size_t)BL0 * 512 * 2);
    float*  hA   = (float*) alloc((size_t)BL0 * 512 * 4);
    float*  hB   = (float*) alloc((size_t)BL0 * 512 * 4);
    float*  xz   = (float*) alloc((size_t)BL0 * 2048 * 4);
    float*  xc   = (float*) alloc((size_t)BL0 * 1024 * 4);
    half_t* xch  = (half_t*)alloc((size_t)BL0 * 1024 * 2);
    float*  proj = (float*) alloc((size_t)BL0 * 64 * 4);
    float*  dtbf = (float*) alloc((size_t)BL0 * 1024 * 4);
    float*  ybuf = (float*) alloc((size_t)BL0 * 1024 * 4);
    half_t* ygh  = (half_t*)alloc((size_t)BL0 * 1024 * 2);
    float*  Pb   = (float*) alloc((size_t)NBATCH * 1024 * 32 * 16 * 4);
    float*  Sb   = (float*) alloc((size_t)NBATCH * 1024 * 32 * 16 * 4);
    float*  H0   = (float*) alloc((size_t)NBATCH * 1024 * 32 * 16 * 4);
    half_t* W_in = (half_t*)alloc((size_t)8 * 2048 * 512 * 2);
    half_t* W_xf = (half_t*)alloc((size_t)8 * 64 * 1024 * 2);
    half_t* W_xb = (half_t*)alloc((size_t)8 * 64 * 1024 * 2);
    half_t* W_dtf= (half_t*)alloc((size_t)8 * 1024 * 32 * 2);
    half_t* W_dtb= (half_t*)alloc((size_t)8 * 1024 * 32 * 2);
    half_t* W_out= (half_t*)alloc((size_t)8 * 512 * 1024 * 2);
    half_t* W_emb= (half_t*)alloc((size_t)2 * 3 * 512 * 512 * 2);

    auto cvt = [&](const float* src, half_t* dst, size_t n) {
        k_f32tof16<<<dim3((unsigned)((n + 255) / 256)), 256, 0, stream>>>(src, dst, n);
    };
    cvt(in_proj_w,  W_in,  (size_t)8 * 2048 * 512);
    cvt(x_proj_f_w, W_xf,  (size_t)8 * 64 * 1024);
    cvt(x_proj_b_w, W_xb,  (size_t)8 * 64 * 1024);
    cvt(dt_proj_f_w,W_dtf, (size_t)8 * 1024 * 32);
    cvt(dt_proj_b_w,W_dtb, (size_t)8 * 1024 * 32);
    cvt(out_proj_w, W_out, (size_t)8 * 512 * 1024);
    {
        size_t n = (size_t)2 * 512 * 512 * 3;
        k_embw<<<dim3((unsigned)((n + 255) / 256)), 256, 0, stream>>>(emb_conv_w, W_emb, n);
    }
    {
        size_t n = (size_t)NBATCH * 512 * L0;
        k_xpose<<<dim3((unsigned)((n + 255) / 256)), 256, 0, stream>>>(x, aTok, n);
    }

    // ---- embedding: two (conv3 as 3 shifted GEMMs) + LN_cf + ReLU ----
    for (int layer = 0; layer < 2; ++layer) {
        dim3 g(512 / 128, BL0 / 64);
        k_gemm<half_t, 0, false><<<g, 128, 0, stream>>>(
            aTok, W_emb + ((size_t)layer * 3 + 0) * 512 * 512, hB,
            BL0, 512, 512, 512, 512, nullptr, nullptr, nullptr, -1, L0);
        k_gemm<half_t, 0, true><<<g, 128, 0, stream>>>(
            aTok, W_emb + ((size_t)layer * 3 + 1) * 512 * 512, hB,
            BL0, 512, 512, 512, 512, nullptr, nullptr, nullptr, 0, L0);
        k_gemm<half_t, 0, true><<<g, 128, 0, stream>>>(
            aTok, W_emb + ((size_t)layer * 3 + 2) * 512 * 512, hB,
            BL0, 512, 512, 512, 512, nullptr, nullptr, nullptr, 1, L0);
        k_ln<true, true, true><<<BL0 / 8, 256, 0, stream>>>(
            hB, emb_ln_w + layer * 512, emb_ln_b + layer * 512, hA, aTok, BL0);
    }

    // ---- bidirectional Mamba blocks ----
    float* hcur = hA;
    float* hoth = hB;
    int Lcur = L0;
    size_t out_off = 0;
    for (int blk = 0; blk < 8; ++blk) {
        int BLc = NBATCH * Lcur;
        int NC = Lcur / SCH;

        k_ln<false, false, true><<<BLc / 8, 256, 0, stream>>>(
            hcur, blk_ln_w + blk * 512, blk_ln_b + blk * 512, nullptr, aTok, BLc);
        {
            dim3 g(2048 / 128, BLc / 64);
            k_gemm<half_t, 0, false><<<g, 128, 0, stream>>>(
                aTok, W_in + (size_t)blk * 2048 * 512, xz,
                BLc, 2048, 512, 512, 2048, nullptr, nullptr, nullptr, 0, Lcur);
        }
        for (int dir = 0; dir < 2; ++dir) {
            const float* cw     = (dir ? conv_b_w : conv_f_w) + (size_t)blk * 1024 * 4;
            const float* cb     = (dir ? conv_b_b : conv_f_b) + (size_t)blk * 1024;
            const half_t* xw    = (dir ? W_xb : W_xf) + (size_t)blk * 64 * 1024;
            const half_t* dw    = (dir ? W_dtb : W_dtf) + (size_t)blk * 1024 * 32;
            const float* dtbias = (dir ? dt_proj_b_b : dt_proj_f_b) + (size_t)blk * 1024;
            const float* Alog   = (dir ? A_log_b : A_log_f) + (size_t)blk * 1024 * 16;
            const float* Dp     = (dir ? D_b : D_f) + (size_t)blk * 1024;

            k_dwconv<<<dim3(4, Lcur / 64, NBATCH), 256, 0, stream>>>(
                xz, cw, cb, xc, xch, Lcur, dir);
            {
                dim3 g(1, BLc / 64);   // N=64 fits one 128-tile
                k_gemm<half_t, 0, false><<<g, 128, 0, stream>>>(
                    xch, xw, proj, BLc, 64, 1024, 1024, 64,
                    nullptr, nullptr, nullptr, 0, Lcur);
            }
            {
                dim3 g(1024 / 128, BLc / 64);
                k_gemm<float, 1, false><<<g, 128, 0, stream>>>(
                    proj, dw, dtbf, BLc, 1024, 32, 64, 1024,
                    dtbias, nullptr, nullptr, 0, Lcur);
            }
            k_scan1<<<dim3(4, NC, NBATCH), 256, 0, stream>>>(
                dtbf, xc, proj, Alog, Pb, Sb, Lcur, NC, dir);
            k_scan2<<<dim3(NBATCH * 1024 * 16 / 256), 256, 0, stream>>>(Pb, Sb, H0, NC);
            k_scan3<<<dim3(4, NC, NBATCH), 256, 0, stream>>>(
                dtbf, xc, proj, Alog, H0, Dp, ybuf, Lcur, NC, dir, dir);
        }
        {
            size_t tot = (size_t)BLc * 1024;
            k_gate<<<dim3((unsigned)((tot + 255) / 256)), 256, 0, stream>>>(xz, ybuf, ygh, tot);
        }
        {
            dim3 g(512 / 128, BLc / 64);
            k_gemm<half_t, 2, false><<<g, 128, 0, stream>>>(
                ygh, W_out + (size_t)blk * 512 * 1024, hcur,
                BLc, 512, 1024, 1024, 512, nullptr, affine + blk * 512, hcur, 0, Lcur);
        }
        if (blk == 3) {
            k_fpn<<<BLc / 8, 256, 0, stream>>>(
                hcur, fpn_ln_w, fpn_ln_b, outp + out_off, Lcur, BLc);
            out_off += (size_t)NBATCH * 512 * Lcur;
        }
        if (blk >= 4) {
            int Lout = Lcur / 2;
            size_t tot = (size_t)NBATCH * Lout * 512;
            k_pool<<<dim3((unsigned)((tot + 255) / 256)), 256, 0, stream>>>(
                hcur, hoth, Lcur, Lout, tot);
            float* t = hcur; hcur = hoth; hoth = t;
            Lcur = Lout;
            int kf = blk - 3;
            k_fpn<<<(NBATCH * Lcur) / 8, 256, 0, stream>>>(
                hcur, fpn_ln_w + kf * 512, fpn_ln_b + kf * 512,
                outp + out_off, Lcur, NBATCH * Lcur);
            out_off += (size_t)NBATCH * 512 * Lcur;
        }
    }
}
